// GCNPooler_4647154614448
// MI455X (gfx1250) — matrix operand order repaired and checked by reading, not compiled, not run on hardware
//
#include <hip/hip_runtime.h>
#include <hip/hip_bf16.h>
#include <math.h>

typedef __attribute__((ext_vector_type(16))) __bf16 v16bf;
typedef __attribute__((ext_vector_type(8)))  __bf16 v8bf;
typedef __attribute__((ext_vector_type(8)))  float  v8f;
typedef __attribute__((ext_vector_type(4)))  float  v4f;

static constexpr int Bsz = 64;
static constexpr int Hd  = 256;
static constexpr int Nn  = 65536;
static constexpr int Ee  = 262144;

// -------- CDNA5 async global->LDS copy (ASYNCcnt path, no VGPR data) -------
__device__ __forceinline__ void async_copy_lds_b128(void* lds_dst, const void* gsrc) {
    unsigned lds_addr = (unsigned)(uintptr_t)lds_dst;   // low 32 bits = LDS byte addr
    asm volatile("global_load_async_to_lds_b128 %0, %1, off"
                 :: "v"(lds_addr), "v"(gsrc) : "memory");
}
__device__ __forceinline__ void wait_async0() {
#if __has_builtin(__builtin_amdgcn_s_wait_asynccnt)
    __builtin_amdgcn_s_wait_asynccnt(0);
#else
    asm volatile("s_wait_asynccnt 0" ::: "memory");
#endif
}

// ---------------------------------------------------------------- utilities
__global__ __launch_bounds__(256) void k_zero(float* __restrict__ p, int n) {
    int i = blockIdx.x * 256 + threadIdx.x;
    if (i < n) p[i] = 0.0f;
}

__global__ __launch_bounds__(256) void k_cvt_bf16(const float* __restrict__ s,
                                                  __bf16* __restrict__ d, int n) {
    int i = blockIdx.x * 256 + threadIdx.x;
    if (i < n) d[i] = (__bf16)s[i];
}

// ------------------------------------------------- 1) gather + filter dots
// one wave (32 lanes) per node: gather 256 floats, store bf16 copy,
// compute a[n]=x·Wf[0:H], b[n]=x·Wf[H:2H], init deg[n]=1 (self loop)
__global__ __launch_bounds__(256) void k_gather(const float* __restrict__ hs,
                                                const int* __restrict__ node_idx,
                                                const float* __restrict__ Wf,
                                                __bf16* __restrict__ xb,
                                                float* __restrict__ av,
                                                float* __restrict__ bv,
                                                float* __restrict__ deg) {
    const int node = (blockIdx.x * 256 + threadIdx.x) >> 5;
    const int lane = threadIdx.x & 31;
    if (node >= Nn) return;
    const int pos = node_idx[node];
    const float* src = hs + (size_t)pos * Hd + lane * 8;
    v4f h0 = *(const v4f*)(src);
    v4f h1 = *(const v4f*)(src + 4);
    v4f wa0 = *(const v4f*)(Wf + lane * 8);
    v4f wa1 = *(const v4f*)(Wf + lane * 8 + 4);
    v4f wb0 = *(const v4f*)(Wf + Hd + lane * 8);
    v4f wb1 = *(const v4f*)(Wf + Hd + lane * 8 + 4);
    float da = 0.f, db = 0.f;
    for (int i = 0; i < 4; ++i) {
        da += h0[i] * wa0[i] + h1[i] * wa1[i];
        db += h0[i] * wb0[i] + h1[i] * wb1[i];
    }
    v8bf o;
    for (int i = 0; i < 4; ++i) { o[i] = (__bf16)h0[i]; o[i + 4] = (__bf16)h1[i]; }
    *(v8bf*)(xb + (size_t)node * Hd + lane * 8) = o;
    for (int off = 16; off >= 1; off >>= 1) {
        da += __shfl_xor(da, off, 32);
        db += __shfl_xor(db, off, 32);
    }
    if (lane == 0) { av[node] = da; bv[node] = db; deg[node] = 1.0f; }
}

// ------------------------------------------- 2) edge filter weight + degree
__global__ __launch_bounds__(256) void k_edge_w(const int* __restrict__ ei,
                                                const float* __restrict__ av,
                                                const float* __restrict__ bv,
                                                const float* __restrict__ bf0,
                                                float* __restrict__ w,
                                                float* __restrict__ deg) {
    int e = blockIdx.x * 256 + threadIdx.x;
    if (e >= Ee) return;
    int r = ei[e], c = ei[Ee + e];
    float z = av[r] + bv[c] + bf0[0];
    float s = 1.0f / (1.0f + __expf(-z));
    w[e] = s;
    unsafeAtomicAdd(&deg[c], s);
}

// ----------------------------------------------------- 3) dis = rsqrt(deg)
__global__ __launch_bounds__(256) void k_dis(float* __restrict__ deg) {
    int i = blockIdx.x * 256 + threadIdx.x;
    if (i < Nn) deg[i] = rsqrtf(deg[i]);   // deg >= 1 always (self loop)
}

// --------------------------------------------- 4) xl = x @ Wc^T  (WMMA GEMM)
// block = 8 waves, computes 64 rows x 256 cols. Wc (bf16) staged in LDS via
// async global->LDS DMA. wave w: row tile mt=w&3, col half jh=w>>2.
__global__ __launch_bounds__(256) void k_gemm_xl(const __bf16* __restrict__ xb,
                                                 const __bf16* __restrict__ wcb,
                                                 float* __restrict__ xl) {
    extern __shared__ unsigned char dyn_smem[];
    __bf16* sW = (__bf16*)dyn_smem;                 // 256x256 bf16 = 128 KB
    for (int i = threadIdx.x; i < Hd * Hd / 8; i += 256)
        async_copy_lds_b128(sW + (size_t)i * 8, wcb + (size_t)i * 8);
    wait_async0();
    __syncthreads();
    const int wid  = threadIdx.x >> 5;
    const int lane = threadIdx.x & 31;
    const int mt = wid & 3, jh = wid >> 2;
    const int lm = lane & 15;
    const int hiA = (lane >> 4) ? 8 : 0;    // A-tile K offset (ISA 16-bit A map)
    const int hiB = (lane >> 4) ? 8 : 0;    // B-tile K offset (ISA 16-bit B map)
    const int row0 = blockIdx.x * 64 + mt * 16;
    v8f acc[8] = {};
    const __bf16* arow = xb + (size_t)(row0 + lm) * Hd;
    for (int kk = 0; kk < Hd; kk += 32) {
        union { v16bf v; v8bf h[2]; } A;
        A.h[0] = *(const v8bf*)(arow + kk + hiA);
        A.h[1] = *(const v8bf*)(arow + kk + hiA + 16);
        for (int jt = 0; jt < 8; ++jt) {
            const __bf16* bp = sW + (size_t)(jh * 128 + jt * 16 + lm) * Hd + kk + hiB;
            union { v16bf v; v8bf h[2]; } Bm;
            Bm.h[0] = *(const v8bf*)(bp);
            Bm.h[1] = *(const v8bf*)(bp + 16);
            acc[jt] = __builtin_amdgcn_wmma_f32_16x16x32_bf16(
                false, A.v, false, Bm.v, (short)0, acc[jt], false, false);
        }
    }
    const int rbase = row0 + ((lane >> 4) ? 8 : 0);   // C map: lanes>=16 hold M=8..15
    for (int jt = 0; jt < 8; ++jt) {
        int j = jh * 128 + jt * 16 + lm;
        for (int r = 0; r < 8; ++r)
            xl[(size_t)(rbase + r) * Hd + j] = acc[jt][r];
    }
}

// ------------------- 5) edge aggregation straight into pooled[64][256] sums
// LDS accumulator per block (64 KB), one wave per edge, 8 h-values per lane
__global__ __launch_bounds__(256) void k_edge_agg(const int* __restrict__ ei,
                                                  const float* __restrict__ w,
                                                  const float* __restrict__ dis,
                                                  const int* __restrict__ nbatch,
                                                  const float* __restrict__ xl,
                                                  float* __restrict__ pooled) {
    extern __shared__ unsigned char dyn_smem[];
    float* acc = (float*)dyn_smem;                  // 64*256 f32 = 64 KB
    for (int i = threadIdx.x; i < Bsz * Hd; i += 256) acc[i] = 0.0f;
    __syncthreads();
    const int lane = threadIdx.x & 31;
    const int wid = (blockIdx.x * 256 + threadIdx.x) >> 5;
    const int nw = (gridDim.x * 256) >> 5;
    for (int e = wid; e < Ee; e += nw) {
        int r = ei[e], c = ei[Ee + e];
        // prefetch next edge's xl row through L2 (global_prefetch_b8)
        if (e + nw < Ee) {
            int rn = ei[e + nw];
            __builtin_prefetch(xl + (size_t)rn * Hd + lane * 8, 0, 0);
        }
        float nrm = dis[r] * w[e] * dis[c];
        int g = nbatch[c];
        const float* xr = xl + (size_t)r * Hd;
        float* ap = acc + g * Hd;
        for (int t = 0; t < 8; ++t) {
            int h = lane + 32 * t;
            atomicAdd(&ap[h], xr[h] * nrm);        // ds_add_f32
        }
    }
    __syncthreads();
    for (int i = threadIdx.x; i < Bsz * Hd; i += 256) {
        float v = acc[i];
        if (v != 0.0f) unsafeAtomicAdd(&pooled[i], v);   // global_atomic_add_f32
    }
}

// ---------------- 6) self-loop term xl[n]*dis[n]^2 + per-batch node counts
__global__ __launch_bounds__(256) void k_self_agg(const float* __restrict__ dis,
                                                  const int* __restrict__ nbatch,
                                                  const float* __restrict__ xl,
                                                  float* __restrict__ pooled,
                                                  float* __restrict__ counts) {
    extern __shared__ unsigned char dyn_smem[];
    float* acc = (float*)dyn_smem;                  // 64*256
    float* cnt = acc + Bsz * Hd;                    // 64
    for (int i = threadIdx.x; i < Bsz * Hd; i += 256) acc[i] = 0.0f;
    if (threadIdx.x < Bsz) cnt[threadIdx.x] = 0.0f;
    __syncthreads();
    const int lane = threadIdx.x & 31;
    const int wid = (blockIdx.x * 256 + threadIdx.x) >> 5;
    const int nw = (gridDim.x * 256) >> 5;
    for (int n = wid; n < Nn; n += nw) {
        float d = dis[n];
        float sn = d * d;                            // 1/deg  (self-loop norm)
        int g = nbatch[n];
        if (lane == 0) atomicAdd(&cnt[g], 1.0f);
        const float* xr = xl + (size_t)n * Hd;
        float* ap = acc + g * Hd;
        for (int t = 0; t < 8; ++t) {
            int h = lane + 32 * t;
            atomicAdd(&ap[h], xr[h] * sn);
        }
    }
    __syncthreads();
    for (int i = threadIdx.x; i < Bsz * Hd; i += 256) {
        float v = acc[i];
        if (v != 0.0f) unsafeAtomicAdd(&pooled[i], v);
    }
    if (threadIdx.x < Bsz) {
        float v = cnt[threadIdx.x];
        if (v != 0.0f) unsafeAtomicAdd(&counts[threadIdx.x], v);
    }
}

// ------------------------- 7) out = tanh((pooled/cnt + bc) @ Wd^T + bd)
// single block, 8 waves; A(64x256) built in LDS, Wd (bf16) async-DMA'd to LDS
__global__ __launch_bounds__(256) void k_final(const float* __restrict__ pooled,
                                               const float* __restrict__ counts,
                                               const float* __restrict__ bc,
                                               const __bf16* __restrict__ wdb,
                                               const float* __restrict__ bd,
                                               float* __restrict__ out) {
    extern __shared__ unsigned char dyn_smem[];
    __bf16* sA = (__bf16*)dyn_smem;                 // 64*256 bf16  = 32 KB
    __bf16* sB = sA + Bsz * Hd;                     // 256*256 bf16 = 128 KB
    for (int i = threadIdx.x; i < Hd * Hd / 8; i += 256)
        async_copy_lds_b128(sB + (size_t)i * 8, wdb + (size_t)i * 8);
    for (int i = threadIdx.x; i < Bsz * Hd; i += 256) {
        int g = i >> 8, h = i & 255;
        float c = counts[g]; if (c < 1.0f) c = 1.0f;
        sA[i] = (__bf16)(pooled[i] / c + bc[h]);
    }
    wait_async0();
    __syncthreads();
    const int wid  = threadIdx.x >> 5;
    const int lane = threadIdx.x & 31;
    const int mt = wid & 3, jh = wid >> 2;
    const int lm = lane & 15;
    const int hiA = (lane >> 4) ? 8 : 0;
    const int hiB = (lane >> 4) ? 8 : 0;
    v8f acc[8] = {};
    const __bf16* arow = sA + (size_t)(mt * 16 + lm) * Hd;
    for (int kk = 0; kk < Hd; kk += 32) {
        union { v16bf v; v8bf h[2]; } A;
        A.h[0] = *(const v8bf*)(arow + kk + hiA);
        A.h[1] = *(const v8bf*)(arow + kk + hiA + 16);
        for (int jt = 0; jt < 8; ++jt) {
            const __bf16* bp = sB + (size_t)(jh * 128 + jt * 16 + lm) * Hd + kk + hiB;
            union { v16bf v; v8bf h[2]; } Bm;
            Bm.h[0] = *(const v8bf*)(bp);
            Bm.h[1] = *(const v8bf*)(bp + 16);
            acc[jt] = __builtin_amdgcn_wmma_f32_16x16x32_bf16(
                false, A.v, false, Bm.v, (short)0, acc[jt], false, false);
        }
    }
    const int rbase = mt * 16 + ((lane >> 4) ? 8 : 0);
    for (int jt = 0; jt < 8; ++jt) {
        int j = jh * 128 + jt * 16 + lm;
        for (int r = 0; r < 8; ++r)
            out[(rbase + r) * Hd + j] = tanhf(acc[jt][r] + bd[j]);
    }
}

// ---------------------------------------------------------------- launcher
extern "C" void kernel_launch(void* const* d_in, const int* in_sizes, int n_in,
                              void* d_out, int out_size, void* d_ws, size_t ws_size,
                              hipStream_t stream) {
    (void)in_sizes; (void)n_in; (void)out_size; (void)ws_size;
    const float* hs    = (const float*)d_in[0];
    const int*   nidx  = (const int*)d_in[1];
    const int*   ei    = (const int*)d_in[2];
    const int*   nbat  = (const int*)d_in[3];
    const float* Wf    = (const float*)d_in[4];
    const float* bf0   = (const float*)d_in[5];
    const float* Wc    = (const float*)d_in[6];
    const float* bc    = (const float*)d_in[7];
    const float* Wd    = (const float*)d_in[8];
    const float* bd    = (const float*)d_in[9];
    float* out = (float*)d_out;

    // workspace carve (256B aligned)
    size_t off = 0;
    auto carve = [&](size_t bytes) -> char* {
        char* p = (char*)d_ws + off;
        off += (bytes + 255) & ~(size_t)255;
        return p;
    };
    __bf16* xb     = (__bf16*)carve((size_t)Nn * Hd * 2);   // 32 MB
    float*  av     = (float*) carve((size_t)Nn * 4);
    float*  bv     = (float*) carve((size_t)Nn * 4);
    float*  deg    = (float*) carve((size_t)Nn * 4);        // becomes dis in place
    float*  w      = (float*) carve((size_t)Ee * 4);
    float*  xl     = (float*) carve((size_t)Nn * Hd * 4);   // 64 MB
    __bf16* wcb    = (__bf16*)carve((size_t)Hd * Hd * 2);
    __bf16* wdb    = (__bf16*)carve((size_t)Hd * Hd * 2);
    float*  pooled = (float*) carve((size_t)Bsz * Hd * 4);
    float*  counts = (float*) carve((size_t)Bsz * 4);

    k_zero<<<(Bsz * Hd + 255) / 256, 256, 0, stream>>>(pooled, Bsz * Hd);
    k_zero<<<1, 256, 0, stream>>>(counts, Bsz);
    k_cvt_bf16<<<(Hd * Hd + 255) / 256, 256, 0, stream>>>(Wc, wcb, Hd * Hd);
    k_cvt_bf16<<<(Hd * Hd + 255) / 256, 256, 0, stream>>>(Wd, wdb, Hd * Hd);

    k_gather<<<Nn * 32 / 256, 256, 0, stream>>>(hs, nidx, Wf, xb, av, bv, deg);
    k_edge_w<<<Ee / 256, 256, 0, stream>>>(ei, av, bv, bf0, w, deg);
    k_dis<<<Nn / 256, 256, 0, stream>>>(deg);

    k_gemm_xl<<<Nn / 64, 256, Hd * Hd * sizeof(__bf16), stream>>>(xb, wcb, xl);

    k_edge_agg<<<256, 256, Bsz * Hd * sizeof(float), stream>>>(ei, w, deg, nbat, xl, pooled);
    k_self_agg<<<128, 256, (Bsz * Hd + Bsz) * sizeof(float), stream>>>(deg, nbat, xl, pooled, counts);

    k_final<<<1, 256, (Bsz * Hd + Hd * Hd) * sizeof(__bf16), stream>>>(
        pooled, counts, bc, wdb, bd, out);
}
